// ParetoStatePredictorMIS_44281112822570
// MI455X (gfx1250) — matrix operand order repaired and checked by reading, not compiled, not run on hardware
//
#include <hip/hip_runtime.h>
#include <cstdint>

typedef __attribute__((ext_vector_type(2)))  float        v2f;
typedef __attribute__((ext_vector_type(8)))  float        v8f;
typedef __attribute__((ext_vector_type(16))) __bf16       v16bf;
typedef __attribute__((ext_vector_type(8)))  unsigned int v8u;

#define F_LN   1
#define F_RELU 2
#define F_BIAS 4
#define F_RES  8

// ---------------------------------------------------------------- helpers

__device__ __forceinline__ unsigned f2bf(float f) {
  __bf16 h = (__bf16)f;
  return (unsigned)__builtin_bit_cast(unsigned short, h);
}

// A-matrix (16x32 bf16) per-lane pack from an LDS row (f32), per ISA 7.12.2:
// lane L: M = L%16; kb = (L<16?0:8); v0..3 -> K = ks*32+kb+2v,+1 ; v4..7 -> K = ks*32+16+kb+...
__device__ __forceinline__ v16bf packA_lds(const float* rowp, int ks) {
  const int lane = threadIdx.x & 31;
  const int kb   = (lane < 16) ? 0 : 8;
  const int b0   = ks * 32 + kb;
  v16bf t;
#pragma unroll
  for (int v = 0; v < 4; ++v) {
    t[2 * v]     = (__bf16)rowp[b0 + 2 * v];
    t[2 * v + 1] = (__bf16)rowp[b0 + 2 * v + 1];
    t[8 + 2 * v] = (__bf16)rowp[b0 + 16 + 2 * v];
    t[9 + 2 * v] = (__bf16)rowp[b0 + 16 + 2 * v + 1];
  }
  return t;
}

// B-matrix (32x16 bf16) loaded from pre-packed weights (one 32B load / lane)
__device__ __forceinline__ v16bf loadBpk(const unsigned* Wpk, int nt, int ks, int KS) {
  const int lane = threadIdx.x & 31;
  const v8u u = *(const v8u*)(Wpk + ((size_t)((nt * KS + ks) * 32 + lane)) * 8);
  return __builtin_bit_cast(v16bf, u);
}

__device__ __forceinline__ v8f wmma_bf16(v16bf a, v16bf b, v8f c) {
  return __builtin_amdgcn_wmma_f32_16x16x32_bf16(false, a, false, b, (short)0, c, false, false);
}
__device__ __forceinline__ v8f wmma_f32(v2f a, v2f b, v8f c) {
  return __builtin_amdgcn_wmma_f32_16x16x4_f32(false, a, false, b, (short)0, c, false, false);
}

// Row-wise LayerNorm of a 16 x W tile held in LDS (one wave, 2 lanes per row)
__device__ void ln_rows16(const float* src, int ss, float* dst, int ds, int W,
                          const float* g, const float* bta,
                          float* sRed, float* sMean, float* sInv) {
  const int lane = threadIdx.x & 31;
  const int r = lane & 15, half = lane >> 4;
  const int hw = W >> 1;
  float s = 0.f;
  for (int c = half * hw; c < half * hw + hw; ++c) s += src[r * ss + c];
  sRed[lane] = s; __syncthreads();
  if (lane < 16) sMean[lane] = (sRed[lane] + sRed[lane + 16]) / (float)W;
  __syncthreads();
  float mu = sMean[r]; s = 0.f;
  for (int c = half * hw; c < half * hw + hw; ++c) { float d = src[r * ss + c] - mu; s += d * d; }
  sRed[lane] = s; __syncthreads();
  if (lane < 16) sInv[lane] = rsqrtf((sRed[lane] + sRed[lane + 16]) / (float)W + 1e-5f);
  __syncthreads();
  for (int i = lane; i < 16 * W; i += 32) {
    int rr = i / W, c = i % W;
    dst[rr * ds + c] = (src[rr * ss + c] - sMean[rr]) * sInv[rr] * g[c] + bta[c];
  }
  __syncthreads();
}

// ---------------------------------------------------------------- weight packing

__global__ void pack_w_k(const float* __restrict__ W, unsigned* __restrict__ out, int O, int I) {
  int total = O * I / 2;
  int t = blockIdx.x * 256 + threadIdx.x;
  if (t >= total) return;
  int KS = I >> 5;
  int v = t & 7;
  int lane = (t >> 3) & 31;
  int rest = t >> 8;
  int ks = rest % KS;
  int nt = rest / KS;
  int n = nt * 16 + (lane & 15);
  int kb = (lane < 16) ? 0 : 8;           // B-layout: lanes 16-31 hold K=8..15,24..31
  int k0 = ks * 32 + kb + 2 * v + ((v >= 4) ? 8 : 0);
  unsigned lo = f2bf(W[(size_t)n * I + k0]);
  unsigned hi = f2bf(W[(size_t)n * I + k0 + 1]);
  out[t] = lo | (hi << 16);
}

__global__ void concat_k(const float* __restrict__ we, const float* __restrict__ wg,
                         float* __restrict__ out) {
  int t = blockIdx.x * 256 + threadIdx.x;
  if (t < 1024) out[t] = we[t];
  else if (t < 2048) out[t] = wg[t - 1024];
}

// ---------------------------------------------------------------- token embedding

__global__ void token_embed_k(const float* __restrict__ n_feat, const float* __restrict__ pos_feat,
                              const float* __restrict__ w1, const float* __restrict__ b1,
                              const float* __restrict__ w2, const float* __restrict__ b2,
                              const float* __restrict__ pw, const float* __restrict__ pb,
                              float* __restrict__ nout) {
  __shared__ float t1[256];
  const int tok = blockIdx.x;
  const int d = threadIdx.x;
  const float* nf = n_feat + (size_t)tok * 6;
#pragma unroll
  for (int part = 0; part < 2; ++part) {
    int j = d + part * 128;
    float accum = 0.f;
    for (int k = 0; k < 3; ++k) {
      float hh = w1[j * 2] * nf[k * 2] + w1[j * 2 + 1] * nf[k * 2 + 1] + b1[j];
      accum += fmaxf(hh, 0.f);
    }
    t1[j] = accum;
  }
  __syncthreads();
  float a = b2[d];
  for (int j = 0; j < 256; ++j) a += t1[j] * w2[d * 256 + j];
  a = fmaxf(a, 0.f);
  const float* pf = pos_feat + (size_t)tok * 10;
  float pp = pb[d];
  for (int p = 0; p < 10; ++p) pp += pf[p] * pw[d * 10 + p];
  nout[(size_t)tok * 128 + d] = a + pp;
}

// ------------------------------------------------- block-0 edge tables (NET=2 distinct rows)

__global__ void edge_tables_k(const float* __restrict__ emb,
                              const float* __restrict__ lng, const float* __restrict__ lnb,
                              const float* __restrict__ we, const float* __restrict__ wg,
                              float* __restrict__ eb0tab, float* __restrict__ g0tab) {
  __shared__ float eln[2][128];
  __shared__ float red[128];
  int d = threadIdx.x;
  for (int t = 0; t < 2; ++t) {
    float x = emb[t * 128 + d];
    red[d] = x; __syncthreads();
    for (int s = 64; s > 0; s >>= 1) { if (d < s) red[d] += red[d + s]; __syncthreads(); }
    float mu = red[0] / 128.f; __syncthreads();
    float df = x - mu; red[d] = df * df; __syncthreads();
    for (int s = 64; s > 0; s >>= 1) { if (d < s) red[d] += red[d + s]; __syncthreads(); }
    float inv = rsqrtf(red[0] / 128.f + 1e-5f); __syncthreads();
    eln[t][d] = df * inv * lng[d] + lnb[d];
  }
  __syncthreads();
  if (d < 16) {
    int t = d >> 3, h = d & 7;
    float se = 0.f, sg = 0.f;
    for (int c = 0; c < 128; ++c) { se += we[h * 128 + c] * eln[t][c]; sg += wg[h * 128 + c] * eln[t][c]; }
    eb0tab[t * 8 + h] = se;
    g0tab[t * 8 + h] = 1.f / (1.f + expf(-sg));
  }
}

__global__ void cent0_k(const int* __restrict__ efeat, const float* __restrict__ g0tab,
                        float* __restrict__ cent0) {
  __shared__ float red[256];
  int bm = blockIdx.x;
  int b = bm >> 8, m = bm & 255;
  float c = (efeat[(size_t)bm * 256 + threadIdx.x] == 1) ? 1.f : 0.f;
  red[threadIdx.x] = c; __syncthreads();
  for (int s = 128; s > 0; s >>= 1) { if (threadIdx.x < s) red[threadIdx.x] += red[threadIdx.x + s]; __syncthreads(); }
  if (threadIdx.x < 8) {
    int h = threadIdx.x;
    float c1 = red[0], c0 = 256.f - c1;
    cent0[((size_t)(b * 8 + h)) * 256 + m] = log1pf(c0 * g0tab[h] + c1 * g0tab[8 + h]);
  }
}

// ---------------------------------------------------------------- generic bf16 WMMA GEMM
// Y[rows,O] = epilogue( maybeLN(X[rows,I]) @ W^T ), W packed. 2 waves/WG, 16 rows/wave.
// KS (= I/32) is a template constant so the A fragments stay register-resident.

template <int KS>
__global__ void gemm16_k(const float* __restrict__ X, int ldx,
                         const unsigned* __restrict__ Wpk,
                         const float* __restrict__ bias,
                         const float* __restrict__ lng, const float* __restrict__ lnb,
                         const float* __restrict__ res, int ldr,
                         float* __restrict__ Y, int ldy,
                         int rows, int O, int flags) {
  constexpr int I  = KS * 32;
  constexpr int SI = I + 4;                   // bank-conflict padding
  extern __shared__ float smem[];
  __shared__ float sRed[2][32], sMean[2][16], sInv[2][16];
  const int wid = threadIdx.x >> 5;
  const int lane = threadIdx.x & 31;
  float* sX = smem + (size_t)wid * 16 * SI;
  const int tile = blockIdx.x * 2 + wid;
  for (int i = lane; i < 16 * I; i += 32) {
    int r = i / I, c = i % I;
    int gm = tile * 16 + r;
    sX[r * SI + c] = (gm < rows) ? X[(size_t)gm * ldx + c] : 0.f;
  }
  __syncthreads();
  if (flags & F_LN)
    ln_rows16(sX, SI, sX, SI, I, lng, lnb, sRed[wid], sMean[wid], sInv[wid]);
  v16bf a[KS];
  {
    const float* rowp = sX + (lane & 15) * SI;
#pragma unroll
    for (int ks = 0; ks < KS; ++ks) a[ks] = packA_lds(rowp, ks);
  }
  const int NT = O >> 4;
  for (int nt = 0; nt < NT; ++nt) {
    if (nt + 1 < NT) __builtin_prefetch(Wpk + ((size_t)((nt + 1) * KS) * 32) * 8, 0, 3);
    v8f acc = {};
#pragma unroll
    for (int ks = 0; ks < KS; ++ks)
      acc = wmma_bf16(a[ks], loadBpk(Wpk, nt, ks, KS), acc);
    const int col = nt * 16 + (lane & 15);
    const float bv = (flags & F_BIAS) ? bias[col] : 0.f;
#pragma unroll
    for (int r = 0; r < 8; ++r) {
      int gm = tile * 16 + r + ((lane < 16) ? 0 : 8);
      if (gm < rows) {
        float v = acc[r] + bv;
        if (flags & F_RELU) v = fmaxf(v, 0.f);
        if (flags & F_RES)  v += res[(size_t)gm * ldr + col];
        Y[(size_t)gm * ldy + col] = v;
      }
    }
  }
}

// ---------------------------------------------------------------- attention (f32 WMMA)
// one wave per (b, h, 16-query tile): scores -> (clip, +Eb, [store Eh0]) -> softmax*cent -> A@V
// IS_FIRST is a template constant: block 0 gathers the Eb table via e_feat and stores Eh0;
// block 1 loads the dense Eb1 buffer. (A runtime flag put scalar-branch ladders in the
// 16x8-iteration epilogue.)

template <int IS_FIRST>
__global__ void attn_k(const float* __restrict__ qkv,
                       const int* __restrict__ efeat,
                       const float* __restrict__ eb0tab,
                       const float* __restrict__ EbBuf,
                       const float* __restrict__ cent,
                       float* __restrict__ EhOut,
                       float* __restrict__ Vo) {
  __shared__ float sS[16 * 260];
  __shared__ float sV[256 * 16];
  __shared__ float sAred[32];
  const int lane = threadIdx.x;
  const int wg = blockIdx.x;                 // b*128 + h*16 + mt
  const int mt = wg & 15;
  const int h = (wg >> 4) & 7;
  const int b = wg >> 7;
  const float scale = 0.25f;                 // 1/sqrt(DK=16)
  const int koff = (lane < 16) ? 0 : 2;
  const int mrow = lane & 15;

  v2f aq[4];
  {
    const float* qp = qkv + ((size_t)(b * 256) + mt * 16 + mrow) * 384 + h * 16;
#pragma unroll
    for (int ks = 0; ks < 4; ++ks) { aq[ks].x = qp[ks * 4 + koff]; aq[ks].y = qp[ks * 4 + koff + 1]; }
  }
  for (int i = lane; i < 4096; i += 32) {
    int n = i >> 4, dk = i & 15;
    sV[n * 16 + dk] = qkv[((size_t)(b * 256) + n) * 384 + 256 + h * 16 + dk];
  }
  // scores tiles
  for (int nt = 0; nt < 16; ++nt) {
    v2f bk[4];
    const float* kp = qkv + ((size_t)(b * 256) + nt * 16 + (lane & 15)) * 384 + 128 + h * 16;
#pragma unroll
    for (int ks = 0; ks < 4; ++ks) { bk[ks].x = kp[ks * 4 + koff]; bk[ks].y = kp[ks * 4 + koff + 1]; }
    v8f c = {};
#pragma unroll
    for (int ks = 0; ks < 4; ++ks) c = wmma_f32(aq[ks], bk[ks], c);
#pragma unroll
    for (int r = 0; r < 8; ++r) {
      int ml = r + ((lane < 16) ? 0 : 8);
      int gm = mt * 16 + ml;
      int nn = nt * 16 + (lane & 15);
      float s = fminf(fmaxf(c[r] * scale, -5.f), 5.f);
      if (IS_FIRST) {
        s += eb0tab[efeat[((size_t)(b * 256) + gm) * 256 + nn] * 8 + h];
        EhOut[(((size_t)(b * 8 + h)) * 256 + gm) * 256 + nn] = s;
      } else {
        s += EbBuf[(((size_t)(b * 8 + h)) * 256 + gm) * 256 + nn];
      }
      sS[ml * 260 + nn] = s;
    }
  }
  __syncthreads();
  // softmax (+ fold in dynamic centrality)
  {
    int r = lane & 15, half = lane >> 4;
    float mx = -1e30f;
    for (int c = half * 128; c < half * 128 + 128; ++c) mx = fmaxf(mx, sS[r * 260 + c]);
    sAred[lane] = mx; __syncthreads();
    mx = fmaxf(sAred[r], sAred[r + 16]);
    float sum = 0.f;
    for (int c = half * 128; c < half * 128 + 128; ++c) sum += __expf(sS[r * 260 + c] - mx);
    __syncthreads();
    sAred[lane] = sum; __syncthreads();
    float inv = 1.f / (sAred[r] + sAred[r + 16]);
    float cf = cent[((size_t)(b * 8 + h)) * 256 + mt * 16 + r] * inv;
    for (int c = half * 128; c < half * 128 + 128; ++c)
      sS[r * 260 + c] = __expf(sS[r * 260 + c] - mx) * cf;
  }
  __syncthreads();
  // A @ V  (K=256 in steps of 4)
  v8f c = {};
  const float* pr = &sS[(lane & 15) * 260];
  for (int ks = 0; ks < 64; ++ks) {
    v2f a, bb;
    a.x  = pr[ks * 4 + koff];               a.y  = pr[ks * 4 + koff + 1];
    bb.x = sV[(ks * 4 + koff) * 16 + (lane & 15)];
    bb.y = sV[(ks * 4 + koff + 1) * 16 + (lane & 15)];
    c = wmma_f32(a, bb, c);
  }
#pragma unroll
  for (int r = 0; r < 8; ++r) {
    int gm = mt * 16 + r + ((lane < 16) ? 0 : 8);
    Vo[((size_t)(b * 256) + gm) * 128 + h * 16 + (lane & 15)] = c[r];
  }
}

// ---------------------------------------------------------------- fused edge stream
// per wave (16 edges of one (b,m) row): rebuild e, edge MLP (bf16 WMMA), LN, 128->16 proj.

__global__ void edge_mega_k(const int* __restrict__ efeat,
                            const float* __restrict__ Eh0,
                            const float* __restrict__ emb,
                            const float* __restrict__ oe,
                            const float* __restrict__ ln2g, const float* __restrict__ ln2b,
                            const unsigned* __restrict__ w1pk,
                            const unsigned* __restrict__ w2pk,
                            const unsigned* __restrict__ projpk,
                            const float* __restrict__ ln1g, const float* __restrict__ ln1b,
                            float* __restrict__ Eb1, float* __restrict__ centAcc) {
  __shared__ float ev[16 * 132];
  __shared__ float xln[16 * 132];
  __shared__ float hid[16 * 260];
  __shared__ float soe[1024];
  __shared__ float semb[256];
  __shared__ float ehv[128];
  __shared__ int sidx[16];
  __shared__ float sRed[32], sMean[16], sInv[16];
  const int lane = threadIdx.x;
  const int wg = blockIdx.x;                  // b*4096 + m*16 + nt16
  const int nt16 = wg & 15;
  const int m = (wg >> 4) & 255;
  const int b = wg >> 12;
  const int n0 = nt16 * 16;

  for (int i = lane; i < 1024; i += 32) soe[i] = oe[i];
  for (int i = lane; i < 256; i += 32) semb[i] = emb[i];
  if (lane < 16) sidx[lane] = efeat[((size_t)(b * 256 + m)) * 256 + n0 + lane];
  for (int i = lane; i < 128; i += 32) {
    int h = i >> 4, t = i & 15;
    ehv[t * 8 + h] = Eh0[(((size_t)(b * 8 + h)) * 256 + m) * 256 + n0 + t];
  }
  __syncthreads();
  // e_vec = emb[idx] + sum_h Eh0 * oe[:,h]
  for (int i = lane; i < 2048; i += 32) {
    int t = i >> 7, d = i & 127;
    float v = semb[sidx[t] * 128 + d];
    const float* op = &soe[d * 8];
    const float* ep = &ehv[t * 8];
#pragma unroll
    for (int h = 0; h < 8; ++h) v += ep[h] * op[h];
    ev[t * 132 + d] = v;
  }
  __syncthreads();
  ln_rows16(ev, 132, xln, 132, 128, ln2g, ln2b, sRed, sMean, sInv);
  // hid = relu(xln @ w1^T)  128 -> 256
  {
    v16bf a[4];
    const float* rowp = xln + (lane & 15) * 132;
#pragma unroll
    for (int ks = 0; ks < 4; ++ks) a[ks] = packA_lds(rowp, ks);
    for (int nt = 0; nt < 16; ++nt) {
      v8f acc = {};
#pragma unroll
      for (int ks = 0; ks < 4; ++ks) acc = wmma_bf16(a[ks], loadBpk(w1pk, nt, ks, 4), acc);
      const int col = nt * 16 + (lane & 15);
#pragma unroll
      for (int r = 0; r < 8; ++r) hid[(r + ((lane < 16) ? 0 : 8)) * 260 + col] = fmaxf(acc[r], 0.f);
    }
  }
  __syncthreads();
  // ev += relu(hid @ w2^T)  256 -> 128
  {
    v16bf a[8];
    const float* rowp = hid + (lane & 15) * 260;
#pragma unroll
    for (int ks = 0; ks < 8; ++ks) a[ks] = packA_lds(rowp, ks);
    for (int nt = 0; nt < 8; ++nt) {
      v8f acc = {};
#pragma unroll
      for (int ks = 0; ks < 8; ++ks) acc = wmma_bf16(a[ks], loadBpk(w2pk, nt, ks, 8), acc);
      const int col = nt * 16 + (lane & 15);
#pragma unroll
      for (int r = 0; r < 8; ++r) ev[(r + ((lane < 16) ? 0 : 8)) * 132 + col] += fmaxf(acc[r], 0.f);
    }
  }
  __syncthreads();
  ln_rows16(ev, 132, xln, 132, 128, ln1g, ln1b, sRed, sMean, sInv);
  // [Eb1 | G1] = xln @ [we1;wg1]^T  128 -> 16
  {
    v16bf a[4];
    const float* rowp = xln + (lane & 15) * 132;
#pragma unroll
    for (int ks = 0; ks < 4; ++ks) a[ks] = packA_lds(rowp, ks);
    v8f acc = {};
#pragma unroll
    for (int ks = 0; ks < 4; ++ks) acc = wmma_bf16(a[ks], loadBpk(projpk, 0, ks, 4), acc);
    const int col = lane & 15;
    float gsum = 0.f;
#pragma unroll
    for (int r = 0; r < 8; ++r) {
      int t = r + ((lane < 16) ? 0 : 8);
      float v = acc[r];
      if (col < 8)
        Eb1[(((size_t)(b * 8 + col)) * 256 + m) * 256 + n0 + t] = v;
      else
        gsum += 1.f / (1.f + expf(-v));
    }
    sRed[lane] = gsum;
    __syncthreads();
    if (lane >= 8 && lane < 16) {
      int h = lane - 8;
      atomicAdd(&centAcc[((size_t)(b * 8 + h)) * 256 + m], sRed[lane] + sRed[lane + 16]);
    }
  }
}

__global__ void cent_fin_k(float* __restrict__ c) {
  int t = blockIdx.x * 256 + threadIdx.x;
  c[t] = log1pf(c[t]);
}

// ---------------------------------------------------------------- head

__global__ void head_reduce_k(const float* __restrict__ n, const float* __restrict__ states,
                              const int* __restrict__ vids,
                              float* __restrict__ nsum, float* __restrict__ sein,
                              float* __restrict__ lv) {
  int b = blockIdx.x, d = threadIdx.x;
  float s1 = 0.f, s2 = 0.f;
  for (int i = 0; i < 256; ++i) {
    float x = n[((size_t)(b * 256) + i) * 128 + d];
    s1 += x; s2 += x * states[b * 256 + i];
  }
  nsum[b * 128 + d] = s1;
  sein[b * 128 + d] = s2;
  lv[b * 128 + d] = n[((size_t)(b * 256) + vids[b]) * 128 + d];
}

__global__ void li1_k(const int* __restrict__ lids, const float* __restrict__ w1,
                      const float* __restrict__ b1, float* __restrict__ li1) {
  int t = blockIdx.x * 256 + threadIdx.x;
  if (t >= 1024) return;
  int b = t >> 7, d = t & 127;
  li1[t] = fmaxf((float)lids[b] * w1[d] + b1[d], 0.f);
}

__global__ void head_final_k(const float* __restrict__ agout, const float* __restrict__ inst,
                             const float* __restrict__ li2, const float* __restrict__ lv,
                             const float* __restrict__ g, const float* __restrict__ bt,
                             const float* __restrict__ pw, const float* __restrict__ pb,
                             float* __restrict__ out) {
  __shared__ float row[128];
  __shared__ float red[128];
  int b = blockIdx.x, d = threadIdx.x;
  float v = agout[b * 128 + d] + inst[b * 128 + d] + li2[b * 128 + d] + lv[b * 128 + d];
  red[d] = v; __syncthreads();
  for (int s = 64; s > 0; s >>= 1) { if (d < s) red[d] += red[d + s]; __syncthreads(); }
  float mu = red[0] / 128.f; __syncthreads();
  float df = v - mu; red[d] = df * df; __syncthreads();
  for (int s = 64; s > 0; s >>= 1) { if (d < s) red[d] += red[d + s]; __syncthreads(); }
  float inv = rsqrtf(red[0] / 128.f + 1e-5f);
  row[d] = df * inv * g[d] + bt[d]; __syncthreads();
  if (d < 2) {
    float acc = pb[d];
    for (int c = 0; c < 128; ++c) acc += row[c] * pw[d * 128 + c];
    out[b * 2 + d] = acc;
  }
}

// ---------------------------------------------------------------- launch

extern "C" void kernel_launch(void* const* d_in, const int* in_sizes, int n_in,
                              void* d_out, int out_size, void* d_ws, size_t ws_size,
                              hipStream_t stream) {
  (void)in_sizes; (void)n_in; (void)out_size; (void)ws_size;
  auto FP = [&](int i) { return (const float*)d_in[i]; };
  auto IP = [&](int i) { return (const int*)d_in[i]; };

  const float* n_feat = FP(0);
  const int* e_feat = IP(1);
  const float* pos_feat = FP(2);
  const int* lids = IP(3);
  const int* vids = IP(4);
  const float* states = FP(5);
  // params pytree, sorted-key order: blocks[0], blocks[1], head, token
  const float *b0_ln_e1_b = FP(6), *b0_ln_e1_g = FP(7), *b0_ln_e2_b = FP(8), *b0_ln_e2_g = FP(9);
  const float *b0_ln_n1_b = FP(10), *b0_ln_n1_g = FP(11), *b0_ln_n2_b = FP(12), *b0_ln_n2_g = FP(13);
  const float *b0_me_w1 = FP(14), *b0_me_w2 = FP(15), *b0_mn_w1 = FP(16), *b0_mn_w2 = FP(17);
  const float *b0_oe = FP(18), *b0_on = FP(19), *b0_we = FP(20), *b0_wg = FP(21), *b0_wqkv = FP(22);
  const float *b1_ln_e1_b = FP(23), *b1_ln_e1_g = FP(24), *b1_ln_n1_b = FP(25), *b1_ln_n1_g = FP(26);
  const float *b1_ln_n2_b = FP(27), *b1_ln_n2_g = FP(28), *b1_mn_w1 = FP(29), *b1_mn_w2 = FP(30);
  const float *b1_on = FP(31), *b1_we = FP(32), *b1_wg = FP(33), *b1_wqkv = FP(34);
  const float *h_ag_b1 = FP(35), *h_ag_b2 = FP(36), *h_ag_w1 = FP(37), *h_ag_w2 = FP(38);
  const float *h_ge_b1 = FP(39), *h_ge_b2 = FP(40), *h_ge_w1 = FP(41), *h_ge_w2 = FP(42);
  const float *h_li_b1 = FP(43), *h_li_b2 = FP(44), *h_li_w1 = FP(45), *h_li_w2 = FP(46);
  const float *h_ln_b = FP(47), *h_ln_g = FP(48), *h_pred_b = FP(49), *h_pred_w = FP(50);
  const float *t_emb = FP(51), *t_l1b = FP(52), *t_l1w = FP(53), *t_l2b = FP(54), *t_l2w = FP(55);
  const float *t_posb = FP(56), *t_posw = FP(57);

  // workspace layout
  float* W = (float*)d_ws;
  size_t off = 0;
  auto A = [&](size_t nfl) { float* p = W + off; off += nfl; return p; };
  float* nbuf   = A(2048 * 128);
  float* qkvb   = A(2048 * 384);
  float* Eh0    = A((size_t)8 * 8 * 256 * 256);
  float* Eb1    = A((size_t)8 * 8 * 256 * 256);
  float* cent0  = A(8 * 8 * 256);
  float* centA  = A(8 * 8 * 256);
  float* Vo     = A(2048 * 128);
  float* tmp256 = A(2048 * 256);
  float* eb0tab = A(16);
  float* g0tab  = A(16);
  float* nsum = A(1024); float* sein = A(1024); float* lv = A(1024);
  float* li1b = A(1024); float* li2b = A(1024);
  float* gehid = A(1024); float* instb = A(1024);
  float* aghid = A(2048); float* agout = A(1024);
  float* wcat = A(2048);
  unsigned* pk_qkv0 = (unsigned*)A(24576);
  unsigned* pk_qkv1 = (unsigned*)A(24576);
  unsigned* pk_on0 = (unsigned*)A(8192);
  unsigned* pk_on1 = (unsigned*)A(8192);
  unsigned* pk_mn1_0 = (unsigned*)A(16384);
  unsigned* pk_mn2_0 = (unsigned*)A(16384);
  unsigned* pk_mn1_1 = (unsigned*)A(16384);
  unsigned* pk_mn2_1 = (unsigned*)A(16384);
  unsigned* pk_me1 = (unsigned*)A(16384);
  unsigned* pk_me2 = (unsigned*)A(16384);
  unsigned* pk_proj = (unsigned*)A(1024);
  unsigned* pk_gew1 = (unsigned*)A(8192);
  unsigned* pk_gew2 = (unsigned*)A(8192);
  unsigned* pk_liw2 = (unsigned*)A(8192);
  unsigned* pk_agw1 = (unsigned*)A(16384);
  unsigned* pk_agw2 = (unsigned*)A(16384);

  hipMemsetAsync(centA, 0, 8 * 8 * 256 * sizeof(float), stream);

  auto PACK = [&](const float* Wsrc, unsigned* dst, int O, int I) {
    int total = O * I / 2;
    pack_w_k<<<(total + 255) / 256, 256, 0, stream>>>(Wsrc, dst, O, I);
  };
  PACK(b0_wqkv, pk_qkv0, 384, 128);
  PACK(b1_wqkv, pk_qkv1, 384, 128);
  PACK(b0_on, pk_on0, 128, 128);
  PACK(b1_on, pk_on1, 128, 128);
  PACK(b0_mn_w1, pk_mn1_0, 256, 128);
  PACK(b0_mn_w2, pk_mn2_0, 128, 256);
  PACK(b1_mn_w1, pk_mn1_1, 256, 128);
  PACK(b1_mn_w2, pk_mn2_1, 128, 256);
  PACK(b0_me_w1, pk_me1, 256, 128);
  PACK(b0_me_w2, pk_me2, 128, 256);
  PACK(h_ge_w1, pk_gew1, 128, 128);
  PACK(h_ge_w2, pk_gew2, 128, 128);
  PACK(h_li_w2, pk_liw2, 128, 128);
  PACK(h_ag_w1, pk_agw1, 256, 128);
  PACK(h_ag_w2, pk_agw2, 128, 256);
  concat_k<<<8, 256, 0, stream>>>(b1_we, b1_wg, wcat);
  PACK(wcat, pk_proj, 16, 128);

  auto GEMM = [&](const float* X, int ldx, const unsigned* Wpk, const float* bias,
                  const float* lng, const float* lnb, const float* res, int ldr,
                  float* Y, int ldy, int rows, int I, int O, int flags) {
    int grid = (rows + 31) / 32;
    size_t sh = (size_t)2 * 16 * (I + 4) * sizeof(float);
    if (I == 128)
      gemm16_k<4><<<grid, 64, sh, stream>>>(X, ldx, Wpk, bias, lng, lnb, res, ldr, Y, ldy, rows, O, flags);
    else
      gemm16_k<8><<<grid, 64, sh, stream>>>(X, ldx, Wpk, bias, lng, lnb, res, ldr, Y, ldy, rows, O, flags);
  };

  token_embed_k<<<2048, 128, 0, stream>>>(n_feat, pos_feat, t_l1w, t_l1b, t_l2w, t_l2b,
                                          t_posw, t_posb, nbuf);
  edge_tables_k<<<1, 128, 0, stream>>>(t_emb, b0_ln_e1_g, b0_ln_e1_b, b0_we, b0_wg, eb0tab, g0tab);
  cent0_k<<<2048, 256, 0, stream>>>(e_feat, g0tab, cent0);

  // ---- block 0
  GEMM(nbuf, 128, pk_qkv0, nullptr, b0_ln_n1_g, b0_ln_n1_b, nullptr, 0,
       qkvb, 384, 2048, 128, 384, F_LN);
  attn_k<1><<<1024, 32, 0, stream>>>(qkvb, e_feat, eb0tab, nullptr, cent0, Eh0, Vo);
  GEMM(Vo, 128, pk_on0, nullptr, nullptr, nullptr, nbuf, 128, nbuf, 128, 2048, 128, 128, F_RES);
  GEMM(nbuf, 128, pk_mn1_0, nullptr, b0_ln_n2_g, b0_ln_n2_b, nullptr, 0,
       tmp256, 256, 2048, 128, 256, F_LN | F_RELU);
  GEMM(tmp256, 256, pk_mn2_0, nullptr, nullptr, nullptr, nbuf, 128,
       nbuf, 128, 2048, 256, 128, F_RELU | F_RES);

  // ---- fused edge stream (produces Eb1, centAcc); 68.7 GFLOP of bf16 WMMA
  edge_mega_k<<<8 * 256 * 16, 32, 0, stream>>>(e_feat, Eh0, t_emb, b0_oe,
                                               b0_ln_e2_g, b0_ln_e2_b,
                                               pk_me1, pk_me2, pk_proj,
                                               b1_ln_e1_g, b1_ln_e1_b, Eb1, centA);
  cent_fin_k<<<64, 256, 0, stream>>>(centA);

  // ---- block 1
  GEMM(nbuf, 128, pk_qkv1, nullptr, b1_ln_n1_g, b1_ln_n1_b, nullptr, 0,
       qkvb, 384, 2048, 128, 384, F_LN);
  attn_k<0><<<1024, 32, 0, stream>>>(qkvb, nullptr, nullptr, Eb1, centA, nullptr, Vo);
  GEMM(Vo, 128, pk_on1, nullptr, nullptr, nullptr, nbuf, 128, nbuf, 128, 2048, 128, 128, F_RES);
  GEMM(nbuf, 128, pk_mn1_1, nullptr, b1_ln_n2_g, b1_ln_n2_b, nullptr, 0,
       tmp256, 256, 2048, 128, 256, F_LN | F_RELU);
  GEMM(tmp256, 256, pk_mn2_1, nullptr, nullptr, nullptr, nbuf, 128,
       nbuf, 128, 2048, 256, 128, F_RELU | F_RES);

  // ---- head
  head_reduce_k<<<8, 128, 0, stream>>>(nbuf, states, vids, nsum, sein, lv);
  li1_k<<<4, 256, 0, stream>>>(lids, h_li_w1, h_li_b1, li1b);
  GEMM(nsum, 128, pk_gew1, h_ge_b1, nullptr, nullptr, nullptr, 0, gehid, 128, 8, 128, 128, F_BIAS | F_RELU);
  GEMM(gehid, 128, pk_gew2, h_ge_b2, nullptr, nullptr, nullptr, 0, instb, 128, 8, 128, 128, F_BIAS | F_RELU);
  GEMM(li1b, 128, pk_liw2, h_li_b2, nullptr, nullptr, nullptr, 0, li2b, 128, 8, 128, 128, F_BIAS | F_RELU);
  GEMM(sein, 128, pk_agw1, h_ag_b1, nullptr, nullptr, nullptr, 0, aghid, 256, 8, 128, 256, F_BIAS | F_RELU);
  GEMM(aghid, 256, pk_agw2, h_ag_b2, nullptr, nullptr, nullptr, 0, agout, 128, 8, 256, 128, F_BIAS | F_RELU);
  head_final_k<<<8, 128, 0, stream>>>(agout, instb, li2b, lv, h_ln_g, h_ln_b,
                                      h_pred_w, h_pred_b, (float*)d_out);
}
